// BiSLSTM_51711406243890
// MI455X (gfx1250) — matrix operand order repaired and checked by reading, not compiled, not run on hardware
//
#include <hip/hip_runtime.h>

// ---------------------------------------------------------------------------
// BiSLSTM for MI455X (gfx1250): bf16 WMMA recurrence, batch-split persistence
// ---------------------------------------------------------------------------
// B=32, S=1024, E=H=256, SHARE=2. 4 persistent workgroups (dir x 16-row batch
// tile), 512 threads = 16 waves each. Weights pre-packed into WMMA B-fragment
// layout (bf16) in d_ws and streamed from L2 every step (anti-LICM barrier
// keeps them inside the time loop); A-fragments staged in LDS per step with
// b128 stores. All recurrent state (h, c, s) lives in LDS (88KB of the WGP's
// 320KB) so no device-scope fences are needed in the sequential loop.

typedef __attribute__((ext_vector_type(16))) __bf16 v16bf;
typedef __attribute__((ext_vector_type(8)))  float  v8f;

#define S_  1024
#define B_  32
#define E_  256
#define H_  256
#define H2_ 512
#define KS_ 8            // K-steps of 32 over a 256-deep contraction

// packed-weight layout offsets (in 16-bit elements), per direction
#define WX_OFF   0
#define WH_OFF   262144
#define WS_OFF   524288
#define WRX_OFF  786432
#define WRH_OFF  917504
#define DIR_STRIDE 1048576

__device__ __forceinline__ unsigned short f2bf(float f) {
    return __builtin_bit_cast(unsigned short, (__bf16)f);   // HW RNE convert
}
__device__ __forceinline__ unsigned int pack_bf2(float a, float b) {
    return (unsigned int)f2bf(a) | ((unsigned int)f2bf(b) << 16);
}
__device__ __forceinline__ float sig_f(float x)  { return 1.f / (1.f + __expf(-x)); }
__device__ __forceinline__ float tanh_f(float x) {
    x = fminf(fmaxf(x, -15.f), 15.f);
    float e = __expf(2.f * x);
    return (e - 1.f) / (e + 1.f);
}

// ---------------------------------------------------------------------------
// Pack an fp32 KxN weight matrix into bf16 WMMA B-fragment order:
//   frag[ntile][kstep][lane][j],  K = kstep*32 + (lane>=16)*16 + j,
//                                 N = ntile*16 + lane%16
// ---------------------------------------------------------------------------
__global__ void pack_wb(const float* __restrict__ src, unsigned short* __restrict__ dst,
                        int Kdim, int Ndim) {
    int p = blockIdx.x * 256 + threadIdx.x;
    if (p >= Kdim * Ndim) return;
    int ks = Kdim >> 5;
    int j  = p & 15;
    int ln = (p >> 4) & 31;
    int q  = p >> 9;
    int kk = q % ks;
    int nt = q / ks;
    int K  = (kk << 5) + ((ln >> 4) << 4) + j;
    int N  = (nt << 4) + (ln & 15);
    dst[p] = f2bf(src[K * Ndim + N]);
}

// ---------------------------------------------------------------------------
// Persistent recurrence kernel. gridDim.x = 4: blockIdx.x = dir*2 + mtile.
// ---------------------------------------------------------------------------
__global__ __launch_bounds__(512)
void bislstm_main(const float* __restrict__ X,        // (B,S,E)
                  const float* __restrict__ mask,     // (B,S)
                  const unsigned short* __restrict__ wfrag,
                  const float* __restrict__ b_f,  const float* __restrict__ br_f,
                  const float* __restrict__ b_r,  const float* __restrict__ br_r,
                  const int*   __restrict__ idxp,
                  float* __restrict__ hOut,           // (S,B,2H)
                  float* __restrict__ cOut,           // (S,B,2H)
                  float* __restrict__ sOut)           // (K,S,B,2H)
{
    __shared__ __attribute__((aligned(16))) unsigned short aX[KS_][32][16]; // 8KB
    __shared__ __attribute__((aligned(16))) unsigned short aH[KS_][32][16]; // 8KB
    __shared__ __attribute__((aligned(16))) unsigned short aS[KS_][32][16]; // 8KB
    __shared__ float hSt[16][H_];                                           // 16KB
    __shared__ float cSt[16][H_];                                           // 16KB
    __shared__ float sSt[2][16][H_];                                        // 32KB

    const int dir  = blockIdx.x >> 1;
    const int mt   = blockIdx.x & 1;
    const int tid  = threadIdx.x;
    const int lane = tid & 31;
    const int wave = tid >> 5;
    const int idx  = idxp[0];

    const unsigned short* wb = wfrag + dir * DIR_STRIDE;
    const float* bz  = dir ? b_r  : b_f;
    const float* brv = dir ? br_r : br_f;
    const int dirOff = dir * H_;

    // fragment-staging coordinates: one 16-byte half-row per thread per matrix
    const int frow  = tid >> 1;          // 0..255  = kk*32 + ln
    const int fhalf = tid & 1;           // low/high 8 halves of the row
    const int fkk   = frow >> 5;
    const int fln   = frow & 31;
    const int fM    = fln & 15;
    const int Kbase = ((fln >> 4) << 4) + (fkk << 5) + (fhalf << 3);
    const int fdst  = (frow << 4) + (fhalf << 3);   // ushort index into frag array

    // zero recurrent state (all LDS)
    for (int i = tid; i < 16 * H_; i += 512) { hSt[i >> 8][i & 255] = 0.f; cSt[i >> 8][i & 255] = 0.f; }
    for (int i = tid; i < 2 * 16 * H_; i += 512) sSt[0][i >> 8][i & 255] = 0.f;
    __syncthreads();

    const int nht  = wave;                 // this wave's 16-column group
    const int hcol = nht * 16 + (lane & 15);

    // loop-invariant biases -> registers (survive the anti-LICM barrier)
    float biasZ[4], biasR[2];
    #pragma unroll
    for (int g = 0; g < 4; ++g) biasZ[g] = bz[g * H_ + hcol];
    #pragma unroll
    for (int k = 0; k < 2; ++k) biasR[k] = brv[k * H_ + hcol];

    for (int t = 0; t < S_; ++t) {
        // opaque barrier: forces weight-fragment loads to stay inside the
        // time loop (otherwise LICM hoists ~1024 VGPRs worth and spills)
        asm volatile("" ::: "memory");

        const int xt = dir ? (S_ - 1 - t) : t;   // x time == output time

        // -------- Phase A: stage A-fragments (x_t, h_prev, s_prev[idx]) ----
        {
            // x_t from global
            const float* srcx = X + ((size_t)(mt * 16 + fM) * S_ + xt) * E_ + Kbase;
            unsigned int w0 = pack_bf2(srcx[0], srcx[1]);
            unsigned int w1 = pack_bf2(srcx[2], srcx[3]);
            unsigned int w2 = pack_bf2(srcx[4], srcx[5]);
            unsigned int w3 = pack_bf2(srcx[6], srcx[7]);
            *(uint4*)&(&aX[0][0][0])[fdst] = make_uint4(w0, w1, w2, w3);
            // h_prev from LDS
            const float* srch = &hSt[fM][Kbase];
            w0 = pack_bf2(srch[0], srch[1]);
            w1 = pack_bf2(srch[2], srch[3]);
            w2 = pack_bf2(srch[4], srch[5]);
            w3 = pack_bf2(srch[6], srch[7]);
            *(uint4*)&(&aH[0][0][0])[fdst] = make_uint4(w0, w1, w2, w3);
            // s_prev[idx] from LDS
            const float* srcs = &sSt[idx][fM][Kbase];
            w0 = pack_bf2(srcs[0], srcs[1]);
            w1 = pack_bf2(srcs[2], srcs[3]);
            w2 = pack_bf2(srcs[4], srcs[5]);
            w3 = pack_bf2(srcs[6], srcs[7]);
            *(uint4*)&(&aS[0][0][0])[fdst] = make_uint4(w0, w1, w2, w3);
        }
        __syncthreads();

        // -------- Phase B: z = x@Wx + h@Wh + s@Ws + b  (4 gate tiles) ------
        v8f acc[4];
        #pragma unroll
        for (int g = 0; g < 4; ++g) {
            float bb = biasZ[g];
            acc[g] = (v8f){bb, bb, bb, bb, bb, bb, bb, bb};
        }
        #pragma unroll
        for (int kk = 0; kk < KS_; ++kk) {
            v16bf ax = *(const v16bf*)&aX[kk][lane][0];
            v16bf ah = *(const v16bf*)&aH[kk][lane][0];
            v16bf as = *(const v16bf*)&aS[kk][lane][0];
            #pragma unroll
            for (int g = 0; g < 4; ++g) {
                int nt = g * 16 + nht;
                const v16bf* bx = (const v16bf*)(wb + WX_OFF + ((size_t)(nt * KS_ + kk) * 32 + lane) * 16);
                const v16bf* bh = (const v16bf*)(wb + WH_OFF + ((size_t)(nt * KS_ + kk) * 32 + lane) * 16);
                const v16bf* bs = (const v16bf*)(wb + WS_OFF + ((size_t)(nt * KS_ + kk) * 32 + lane) * 16);
                acc[g] = __builtin_amdgcn_wmma_f32_16x16x32_bf16(false, ax, false, *bx, (short)0, acc[g], false, false);
                acc[g] = __builtin_amdgcn_wmma_f32_16x16x32_bf16(false, ah, false, *bh, (short)0, acc[g], false, false);
                acc[g] = __builtin_amdgcn_wmma_f32_16x16x32_bf16(false, as, false, *bs, (short)0, acc[g], false, false);
            }
        }

        // -------- Phase C: LSTM gates, masked state update, h/c outputs ----
        float mv[8];
        #pragma unroll
        for (int v = 0; v < 8; ++v) {
            int m = v + ((lane >> 4) << 3);      // C layout: M = v + 8*(lane>=16)
            mv[v] = mask[(size_t)(mt * 16 + m) * S_ + xt];
        }
        #pragma unroll
        for (int v = 0; v < 8; ++v) {
            int   m  = v + ((lane >> 4) << 3);
            int   bg = mt * 16 + m;
            float cn = sig_f(acc[1][v]) * cSt[m][hcol] + sig_f(acc[0][v]) * tanh_f(acc[2][v]);
            float hn = sig_f(acc[3][v]) * tanh_f(cn);
            float hm = mv[v] * hn + (1.f - mv[v]) * hSt[m][hcol];
            float cm = mv[v] * cn + (1.f - mv[v]) * cSt[m][hcol];
            hSt[m][hcol] = hm;
            cSt[m][hcol] = cm;
            hOut[((size_t)xt * B_ + bg) * H2_ + dirOff + hcol] = hm;
            cOut[((size_t)xt * B_ + bg) * H2_ + dirOff + hcol] = cm;
        }
        __syncthreads();

        // -------- Phase D: rebuild h A-fragments from updated hSt ----------
        {
            const float* srch = &hSt[fM][Kbase];
            unsigned int w0 = pack_bf2(srch[0], srch[1]);
            unsigned int w1 = pack_bf2(srch[2], srch[3]);
            unsigned int w2 = pack_bf2(srch[4], srch[5]);
            unsigned int w3 = pack_bf2(srch[6], srch[7]);
            *(uint4*)&(&aH[0][0][0])[fdst] = make_uint4(w0, w1, w2, w3);
        }
        __syncthreads();

        // -------- Phase E: r = sig(x@Wrx + h@Wrh + br); s update -----------
        #pragma unroll
        for (int k = 0; k < 2; ++k) {
            float bb = biasR[k];
            v8f racc = (v8f){bb, bb, bb, bb, bb, bb, bb, bb};
            #pragma unroll
            for (int kk = 0; kk < KS_; ++kk) {
                v16bf ax = *(const v16bf*)&aX[kk][lane][0];
                v16bf ah = *(const v16bf*)&aH[kk][lane][0];
                const v16bf* bx = (const v16bf*)(wb + WRX_OFF + k * 65536 + ((size_t)(nht * KS_ + kk) * 32 + lane) * 16);
                const v16bf* bh = (const v16bf*)(wb + WRH_OFF + k * 65536 + ((size_t)(nht * KS_ + kk) * 32 + lane) * 16);
                racc = __builtin_amdgcn_wmma_f32_16x16x32_bf16(false, ax, false, *bx, (short)0, racc, false, false);
                racc = __builtin_amdgcn_wmma_f32_16x16x32_bf16(false, ah, false, *bh, (short)0, racc, false, false);
            }
            #pragma unroll
            for (int v = 0; v < 8; ++v) {
                int   m  = v + ((lane >> 4) << 3);
                int   bg = mt * 16 + m;
                float r  = sig_f(racc[v]);
                float so = sSt[k][m][hcol];
                float sn = r * so + (1.f - r) * cSt[m][hcol];
                float sm = mv[v] * sn + (1.f - mv[v]) * so;
                sSt[k][m][hcol] = sm;
                sOut[(((size_t)k * S_ + xt) * B_ + bg) * H2_ + dirOff + hcol] = sm;
            }
        }
        __syncthreads();   // sSt/hSt/cSt stable before next step's staging
    }
}

// ---------------------------------------------------------------------------
extern "C" void kernel_launch(void* const* d_in, const int* in_sizes, int n_in,
                              void* d_out, int out_size, void* d_ws, size_t ws_size,
                              hipStream_t stream) {
    (void)in_sizes; (void)n_in; (void)out_size; (void)ws_size;

    const float* X    = (const float*)d_in[0];
    const float* mask = (const float*)d_in[1];
    const int*   idxp = (const int*)d_in[16];

    unsigned short* wf = (unsigned short*)d_ws;    // 2*1048576 bf16 = 4MB

    // pack weights (both directions) into WMMA B-fragment bf16 layout
    for (int d = 0; d < 2; ++d) {
        const float* Wx  = (const float*)d_in[2 + 7 * d];
        const float* Wh  = (const float*)d_in[3 + 7 * d];
        const float* Ws  = (const float*)d_in[4 + 7 * d];
        const float* Wrx = (const float*)d_in[6 + 7 * d];
        const float* Wrh = (const float*)d_in[7 + 7 * d];
        unsigned short* base = wf + (size_t)d * DIR_STRIDE;
        pack_wb<<<1024, 256, 0, stream>>>(Wx, base + WX_OFF, 256, 1024);
        pack_wb<<<1024, 256, 0, stream>>>(Wh, base + WH_OFF, 256, 1024);
        pack_wb<<<1024, 256, 0, stream>>>(Ws, base + WS_OFF, 256, 1024);
        for (int k = 0; k < 2; ++k) {
            pack_wb<<<256, 256, 0, stream>>>(Wrx + (size_t)k * 65536, base + WRX_OFF + k * 65536, 256, 256);
            pack_wb<<<256, 256, 0, stream>>>(Wrh + (size_t)k * 65536, base + WRH_OFF + k * 65536, 256, 256);
        }
    }

    float* hOut = (float*)d_out;                              // S*B*2H
    float* cOut = hOut + (size_t)S_ * B_ * H2_;               // S*B*2H
    float* sOut = cOut + (size_t)S_ * B_ * H2_;               // K*S*B*2H

    bislstm_main<<<4, 512, 0, stream>>>(
        X, mask, wf,
        (const float*)d_in[5],  (const float*)d_in[8],
        (const float*)d_in[12], (const float*)d_in[15],
        idxp, hOut, cOut, sOut);
}
